// GraphAttentionLayer_42502996362087
// MI455X (gfx1250) — matrix operand order repaired and checked by reading, not compiled, not run on hardware
//
#include <hip/hip_runtime.h>
#include <hip/hip_bf16.h>

#define N_TOK 2048
#define DM    1024
#define NH    16
#define DH    64

typedef __attribute__((ext_vector_type(4)))  __bf16 v4bf;
typedef __attribute__((ext_vector_type(8)))  __bf16 v8bf;
typedef __attribute__((ext_vector_type(16))) __bf16 v16bf;
typedef __attribute__((ext_vector_type(4)))  float  v4f;
typedef __attribute__((ext_vector_type(8)))  float  v8f;
typedef __attribute__((ext_vector_type(4)))  int    v4i;

typedef __attribute__((address_space(1))) v4i* gv4i_ptr;   // global int4*
typedef __attribute__((address_space(3))) v4i* lv4i_ptr;   // LDS int4*

#if __has_builtin(__builtin_amdgcn_global_load_async_to_lds_b128)
#define HAVE_ASYNC_LDS 1
#else
#define HAVE_ASYNC_LDS 0
#endif

static __device__ inline void wait_async_le4() {
#if HAVE_ASYNC_LDS
#if __has_builtin(__builtin_amdgcn_s_wait_asynccnt)
  __builtin_amdgcn_s_wait_asynccnt(4);
#else
  asm volatile("s_wait_asynccnt 0x4" ::: "memory");
#endif
#endif
}
static __device__ inline void wait_async_le0() {
#if HAVE_ASYNC_LDS
#if __has_builtin(__builtin_amdgcn_s_wait_asynccnt)
  __builtin_amdgcn_s_wait_asynccnt(0);
#else
  asm volatile("s_wait_asynccnt 0x0" ::: "memory");
#endif
#endif
}

// Build a 16-element bf16 A/B fragment from two 16-byte runs.
static __device__ inline v16bf ld16(const __bf16* p0, const __bf16* p1) {
  v8bf a = *(const v8bf*)p0;
  v8bf b = *(const v8bf*)p1;
  return __builtin_shufflevector(a, b, 0,1,2,3,4,5,6,7,8,9,10,11,12,13,14,15);
}

static __device__ inline v8f wmma_bf16(v16bf a, v16bf b, v8f c) {
  // (neg_a, A, neg_b, B, c_mod, C, reuse_a, reuse_b)
  return __builtin_amdgcn_wmma_f32_16x16x32_bf16(false, a, false, b, (short)0, c,
                                                 false, false);
}

// ---------------------------------------------------------------- fp32 -> bf16
__global__ __launch_bounds__(256) void cvt_f32_to_bf16(const float* __restrict__ s,
                                                       __bf16* __restrict__ d, int n4) {
  int i = blockIdx.x * 256 + threadIdx.x;
  if (i < n4) {
    v4f x = *(const v4f*)(s + (size_t)i * 4);
    v4bf y;
    y[0] = (__bf16)x[0]; y[1] = (__bf16)x[1];
    y[2] = (__bf16)x[2]; y[3] = (__bf16)x[3];
    *(v4bf*)(d + (size_t)i * 4) = y;
  }
}

// ------------------------------------------------- generic bf16 WMMA GEMM
// out = (A[M,K] @ W[Nn,K]^T + bias) * alpha
// MODE 0: bf16 row-major   MODE 1: bf16 transposed (out[n*M+m])   MODE 2: fp32
template <int MODE>
__global__ __launch_bounds__(256) void gemm_bf16_wmma(
    const __bf16* __restrict__ A, const __bf16* __restrict__ W,
    const float* __restrict__ bias, float alpha, void* __restrict__ outp,
    int M, int Nn, int K) {
  const int tid  = threadIdx.x;
  const int lane = tid & 31, wave = tid >> 5;
  const int l16  = lane & 15, half = lane >> 4;
  const int wrow = wave >> 1, wcol = wave & 1;
  const int m_w  = blockIdx.y * 128 + wrow * 32;
  const int n_w  = blockIdx.x * 128 + wcol * 64;

  v8f acc[2][4] = {};

  for (int kk = 0; kk < K; kk += 32) {
    v16bf af[2], bf[4];
#pragma unroll
    for (int s = 0; s < 2; ++s) {
      const __bf16* p = A + (size_t)(m_w + s * 16 + l16) * K + kk;
      af[s] = ld16(p + 16 * half, p + 16 * half + 8);
    }
#pragma unroll
    for (int t = 0; t < 4; ++t) {
      const __bf16* p = W + (size_t)(n_w + t * 16 + l16) * K + kk;
      bf[t] = ld16(p + 16 * half, p + 16 * half + 8);
    }
#pragma unroll
    for (int s = 0; s < 2; ++s)
#pragma unroll
      for (int t = 0; t < 4; ++t)
        acc[s][t] = wmma_bf16(af[s], bf[t], acc[s][t]);
  }

  __bf16* outB = (__bf16*)outp;
  float*  outF = (float*)outp;
#pragma unroll
  for (int t = 0; t < 4; ++t) {
    const int n = n_w + t * 16 + l16;
    const float bn = bias[n];
#pragma unroll
    for (int s = 0; s < 2; ++s) {
#pragma unroll
      for (int r = 0; r < 8; ++r) {
        const int row = m_w + s * 16 + r + 8 * half;
        float v = (acc[s][t][r] + bn) * alpha;
        if (MODE == 2)      outF[(size_t)row * Nn + n] = v;
        else if (MODE == 1) outB[(size_t)n * M + row]  = (__bf16)v;
        else                outB[(size_t)row * Nn + n] = (__bf16)v;
      }
    }
  }
}

// ------------------------------------------------- fused flash attention
// 512 threads = 16 waves; wave h handles head h for a 16-query tile.
// Edge-bias tiles (the only HBM-bound stream) are staged to LDS with async
// copies, double-buffered so tile t+1 fetch overlaps tile t compute.
__global__ __launch_bounds__(512) void attn_fused(
    const __bf16* __restrict__ qb,   // [N, DM], pre-scaled by 1/sqrt(DH)
    const __bf16* __restrict__ kb,   // [N, DM]
    const __bf16* __restrict__ vT,   // [DM, N]
    const float* __restrict__ eb,    // [N, N, NH]
    __bf16* __restrict__ ob) {       // [N, DM]
  __shared__ float  sb[2][16 * 32 * NH];  // bias ping-pong, raw [i(16)][j(32)][h]
  __shared__ __bf16 sp[NH * 16 * 32];     // per-wave P scratch, [h][row][k]

  const int tid  = threadIdx.x;
  const int lane = tid & 31, h = tid >> 5;
  const int l16  = lane & 15, half = lane >> 4;
  const int i0   = blockIdx.x * 16;

  // stage one 16x32x16 bias tile: 4 x b128 per thread, coalesced
  auto stage = [&](int j0, int buf) {
#pragma unroll
    for (int it = 0; it < 4; ++it) {
      int q4 = it * 512 + tid;  // 2048 float4s
      int i  = q4 >> 7;
      int rm = q4 & 127;
      int j  = rm >> 2;
      int h4 = rm & 3;
      const float* gp = eb + ((size_t)(i0 + i) * N_TOK + (j0 + j)) * NH + h4 * 4;
#if HAVE_ASYNC_LDS
      __builtin_amdgcn_global_load_async_to_lds_b128(
          (gv4i_ptr)gp, (lv4i_ptr)&sb[buf][q4 * 4], 0, 0);
#else
      *(v4f*)&sb[buf][q4 * 4] = *(const v4f*)gp;
      __builtin_prefetch(gp + (size_t)32 * NH, 0, 1);
#endif
    }
  };

  // Q fragments (16x64 = two 16x32 A-fragments)
  v16bf aq[2];
  {
    const __bf16* p = qb + (size_t)(i0 + l16) * DM + h * DH;
    aq[0] = ld16(p + 16 * half, p + 16 * half + 8);
    aq[1] = ld16(p + 32 + 16 * half, p + 40 + 16 * half);
  }

  v8f accO[4] = {};
  float mrow[8], lrow[8];
#pragma unroll
  for (int r = 0; r < 8; ++r) { mrow[r] = -1e30f; lrow[r] = 0.f; }

  stage(0, 0);

  const int nchunk = N_TOK / 32;  // 64
  for (int c = 0; c < nchunk; ++c) {
    const int j0 = c * 32;
    if (c + 1 < nchunk) {
      stage(j0 + 32, (c + 1) & 1);  // overlap next fetch with this compute
      wait_async_le4();             // previous 4 async copies (this tile) done
    } else {
      wait_async_le0();
    }
    __syncthreads();  // tile visible to all waves
    const float* sbc = &sb[c & 1][0];

    // scores for a 32-key chunk: two 16x16 tiles
    v8f s0 = {}, s1 = {};
    {
      const __bf16* p = kb + (size_t)(j0 + l16) * DM + h * DH;
      s0 = wmma_bf16(aq[0], ld16(p + 16 * half, p + 16 * half + 8), s0);
      s0 = wmma_bf16(aq[1], ld16(p + 32 + 16 * half, p + 40 + 16 * half), s0);
    }
    {
      const __bf16* p = kb + (size_t)(j0 + 16 + l16) * DM + h * DH;
      s1 = wmma_bf16(aq[0], ld16(p + 16 * half, p + 16 * half + 8), s1);
      s1 = wmma_bf16(aq[1], ld16(p + 32 + 16 * half, p + 40 + 16 * half), s1);
    }

    // bias add + online softmax (rows live at r + 8*half, cols across 16 lanes)
    float fac[8];
#pragma unroll
    for (int r = 0; r < 8; ++r) {
      const int irow = r + 8 * half;
      float x0 = s0[r] + sbc[(irow * 32 + l16) * NH + h];
      float x1 = s1[r] + sbc[(irow * 32 + 16 + l16) * NH + h];
      float t = fmaxf(x0, x1);
      t = fmaxf(t, __shfl_xor(t, 1, 32));
      t = fmaxf(t, __shfl_xor(t, 2, 32));
      t = fmaxf(t, __shfl_xor(t, 4, 32));
      t = fmaxf(t, __shfl_xor(t, 8, 32));
      float mnew = fmaxf(mrow[r], t);
      float f  = __expf(mrow[r] - mnew);
      float p0 = __expf(x0 - mnew);
      float p1 = __expf(x1 - mnew);
      float ss = p0 + p1;
      ss += __shfl_xor(ss, 1, 32);
      ss += __shfl_xor(ss, 2, 32);
      ss += __shfl_xor(ss, 4, 32);
      ss += __shfl_xor(ss, 8, 32);
      lrow[r] = lrow[r] * f + ss;
      mrow[r] = mnew;
      fac[r]  = f;
      // stash P (C-layout) to wave-private LDS for A-layout reload
      sp[h * 512 + irow * 32 + l16]      = (__bf16)p0;
      sp[h * 512 + irow * 32 + 16 + l16] = (__bf16)p1;
    }

#pragma unroll
    for (int cc = 0; cc < 4; ++cc)
#pragma unroll
      for (int r = 0; r < 8; ++r)
        accO[cc][r] *= fac[r];

    // reload P as a 16x32 bf16 A-fragment (wave-private; LDS in-order in wave)
    const __bf16* pp = &sp[h * 512 + l16 * 32];
    v16bf ap = ld16(pp + 16 * half, pp + 16 * half + 8);

    // P @ V : V^T rows are contiguous along keys -> clean B-fragments
#pragma unroll
    for (int cc = 0; cc < 4; ++cc) {
      const __bf16* p =
          vT + (size_t)(h * DH + cc * 16 + l16) * N_TOK + j0 + 16 * half;
      accO[cc] = wmma_bf16(ap, ld16(p, p + 8), accO[cc]);
    }
    __syncthreads();  // all waves done with tile before next overwrite
  }

  // normalize and store bf16 output rows
#pragma unroll
  for (int cc = 0; cc < 4; ++cc)
#pragma unroll
    for (int r = 0; r < 8; ++r) {
      const int row = i0 + r + 8 * half;
      float o = accO[cc][r] / lrow[r];
      ob[(size_t)row * DM + h * DH + cc * 16 + l16] = (__bf16)o;
    }
}

// ---------------------------------------------------------------- launcher
extern "C" void kernel_launch(void* const* d_in, const int* in_sizes, int n_in,
                              void* d_out, int out_size, void* d_ws, size_t ws_size,
                              hipStream_t stream) {
  (void)in_sizes; (void)n_in; (void)out_size; (void)ws_size;
  const float* x  = (const float*)d_in[0];
  const float* eb = (const float*)d_in[1];
  const float* Wq = (const float*)d_in[2];
  const float* bq = (const float*)d_in[3];
  const float* Wk = (const float*)d_in[4];
  const float* bk = (const float*)d_in[5];
  const float* Wv = (const float*)d_in[6];
  const float* bv = (const float*)d_in[7];
  const float* Wo = (const float*)d_in[8];
  const float* bo = (const float*)d_in[9];
  float* out = (float*)d_out;

  char* ws = (char*)d_ws;                       // 28 MiB used
  __bf16* xb  = (__bf16*)(ws + (0u  << 20));    // [N, DM]       4 MiB
  __bf16* Wqb = (__bf16*)(ws + (4u  << 20));    // [DM, DM]      2 MiB
  __bf16* Wkb = (__bf16*)(ws + (6u  << 20));
  __bf16* Wvb = (__bf16*)(ws + (8u  << 20));
  __bf16* Wob = (__bf16*)(ws + (10u << 20));
  __bf16* qb  = (__bf16*)(ws + (12u << 20));    // [N, DM]       4 MiB
  __bf16* kbf = (__bf16*)(ws + (16u << 20));    // [N, DM]       4 MiB
  __bf16* vTb = (__bf16*)(ws + (20u << 20));    // [DM, N]       4 MiB
  __bf16* aob = (__bf16*)(ws + (24u << 20));    // [N, DM]       4 MiB

  {
    int n4 = (N_TOK * DM) / 4;
    cvt_f32_to_bf16<<<n4 / 256, 256, 0, stream>>>(x, xb, n4);
  }
  {
    int n4 = (DM * DM) / 4;
    cvt_f32_to_bf16<<<n4 / 256, 256, 0, stream>>>(Wq, Wqb, n4);
    cvt_f32_to_bf16<<<n4 / 256, 256, 0, stream>>>(Wk, Wkb, n4);
    cvt_f32_to_bf16<<<n4 / 256, 256, 0, stream>>>(Wv, Wvb, n4);
    cvt_f32_to_bf16<<<n4 / 256, 256, 0, stream>>>(Wo, Wob, n4);
  }

  dim3 gg(DM / 128, N_TOK / 128);  // (8, 16)
  const float scale = 0.125f;      // 1/sqrt(DH)
  gemm_bf16_wmma<0><<<gg, 256, 0, stream>>>(xb, Wqb, bq, scale, qb,
                                            N_TOK, DM, DM);
  gemm_bf16_wmma<0><<<gg, 256, 0, stream>>>(xb, Wkb, bk, 1.0f, kbf,
                                            N_TOK, DM, DM);
  gemm_bf16_wmma<1><<<gg, 256, 0, stream>>>(xb, Wvb, bv, 1.0f, vTb,
                                            N_TOK, DM, DM);   // store V^T

  attn_fused<<<N_TOK / 16, 512, 0, stream>>>(qb, kbf, vTb, eb, aob);

  gemm_bf16_wmma<2><<<gg, 256, 0, stream>>>(aob, Wob, bo, 1.0f, out,
                                            N_TOK, DM, DM);
}
